// ManiPointSegment_44169443672117
// MI455X (gfx1250) — matrix operand order repaired and checked by reading, not compiled, not run on hardware
//
#include <hip/hip_runtime.h>
#include <hip/hip_bf16.h>

typedef __attribute__((ext_vector_type(16))) _Float16 v16h;
typedef __attribute__((ext_vector_type(8)))  _Float16 v8h;
typedef __attribute__((ext_vector_type(8)))  float    v8f;

#define GSTR 40   // LDS row stride in halves (80B: keeps 8-half chunks 16B-aligned)

// ---------------------------------------------------------------------------
// point load helper: layout 0 = row-major (B,N,3); layout 1 = strided (B,3,N)
// ---------------------------------------------------------------------------
__device__ __forceinline__ void load_pt(const float* __restrict__ xyz, int layout,
                                        int b, int n, int N,
                                        float& x, float& y, float& z) {
  if (layout == 0) {
    const float* p = xyz + ((size_t)b * N + n) * 3;
    x = p[0]; y = p[1]; z = p[2];
  } else {
    const float* p = xyz + (size_t)b * 3 * N + n;
    x = p[0]; y = p[N]; z = p[2 * N];
  }
}

// build v16h fragment from two aligned 8-half LDS chunks
__device__ __forceinline__ v16h frag16(const _Float16* p0, const _Float16* p1) {
  v8h lo = *(const v8h*)p0;
  v8h hi = *(const v8h*)p1;
  return __builtin_shufflevector(lo, hi, 0, 1, 2, 3, 4, 5, 6, 7,
                                 8, 9, 10, 11, 12, 13, 14, 15);
}

// ---------------------------------------------------------------------------
// Generic WMMA GEMM: C[M,N] = epi(A[M,K] * W[N,K]^T)
// epi: v += bias (flags&4); v = v*gamma+beta (flags&1); v = relu(v) (flags&2)
// 64x64 tile / workgroup, 256 threads = 8 waves, each wave: 1 Mtile x 2 Ntiles
// ---------------------------------------------------------------------------
__global__ void __launch_bounds__(256) wmma_gemm(
    const float* __restrict__ A, const float* __restrict__ W,
    const float* __restrict__ bias, const float* __restrict__ gamma,
    const float* __restrict__ beta, float* __restrict__ C,
    int M, int N, int K, int flags)
{
  __shared__ _Float16 As[64 * GSTR];
  __shared__ _Float16 Bs[64 * GSTR];
  const int tid  = threadIdx.x;
  const int wave = tid >> 5, lane = tid & 31;
  const int lrow = lane & 15, lhi = lane >> 4;
  const int m0 = blockIdx.y * 64, n0 = blockIdx.x * 64;
  const int mt  = wave & 3;
  const int nt0 = (wave >> 2) * 2;

  // staging assignment: each thread owns one 8-half chunk of one LDS row
  const int row  = tid >> 2;
  const int kloc = (tid & 3) * 8;
  const bool arow = (m0 + row) < M;
  const bool wrow = (n0 + row) < N;
  const float* Ap = A + (size_t)(m0 + row) * K;
  const float* Wp = W + (size_t)(n0 + row) * K;

  v8f acc0 = {}; v8f acc1 = {};

  for (int kb = 0; kb < K; kb += 32) {
    const int kk = kb + kloc;
    v8h ah = {}; v8h wh = {};
    if (kk + 8 <= K) {                       // full chunk
      if (arow) {
#pragma unroll
        for (int j = 0; j < 8; ++j) ah[j] = (_Float16)Ap[kk + j];
      }
      if (wrow) {
#pragma unroll
        for (int j = 0; j < 8; ++j) wh[j] = (_Float16)Wp[kk + j];
      }
    } else if (kk < K) {                     // ragged tail chunk
#pragma unroll
      for (int j = 0; j < 8; ++j) {
        if (kk + j < K) {
          if (arow) ah[j] = (_Float16)Ap[kk + j];
          if (wrow) wh[j] = (_Float16)Wp[kk + j];
        }
      }
    }
    *(v8h*)(&As[row * GSTR + kloc]) = ah;    // ds_store_b128
    *(v8h*)(&Bs[row * GSTR + kloc]) = wh;
    if (kb + 32 < K) {                       // prefetch next chunk
      __builtin_prefetch(Ap + kb + 32 + kloc, 0, 1);
      __builtin_prefetch(Wp + kb + 32 + kloc, 0, 1);
    }
    __syncthreads();

    // ---- build fragments per ISA register layouts (aligned b128 LDS reads) --
    const int ab  = (mt * 16 + lrow) * GSTR;
    const int k0l = lhi ? 8 : 0;
    v16h af = frag16(&As[ab + k0l], &As[ab + 16 + k0l]);

    const int bb0 = (nt0 * 16 + lrow) * GSTR + (lhi ? 8 : 0);
    const int bb1 = bb0 + 16 * GSTR;
    v16h bf0 = frag16(&Bs[bb0], &Bs[bb0 + 16]);
    v16h bf1 = frag16(&Bs[bb1], &Bs[bb1 + 16]);

    acc0 = __builtin_amdgcn_wmma_f32_16x16x32_f16(false, af, false, bf0,
                                                  (short)0, acc0, false, false);
    acc1 = __builtin_amdgcn_wmma_f32_16x16x32_f16(false, af, false, bf1,
                                                  (short)0, acc1, false, false);
    __syncthreads();
  }

  // ---- epilogue + store ----
#pragma unroll
  for (int h = 0; h < 2; ++h) {
    v8f acc = h ? acc1 : acc0;
    const int gn = n0 + (nt0 + h) * 16 + lrow;
    if (gn >= N) continue;
    const float bb = (flags & 4) ? bias[gn]  : 0.f;
    const float gg = (flags & 1) ? gamma[gn] : 1.f;
    const float ee = (flags & 1) ? beta[gn]  : 0.f;
#pragma unroll
    for (int i = 0; i < 8; ++i) {
      const int gm = m0 + mt * 16 + lhi * 8 + i;
      if (gm < M) {
        float v = acc[i] + bb;
        v = v * gg + ee;
        if (flags & 2) v = fmaxf(v, 0.f);
        C[(size_t)gm * N + gn] = v;
      }
    }
  }
}

// ---------------------------------------------------------------------------
// agg[b,s,c,w] = sum_k (h[b,s,k,c]*ds[b,s,k]) * wf[b,s,k,w]
// one workgroup (128 thr = 4 waves) per (b,s); M=Cout, N=16, K=Ks
// ---------------------------------------------------------------------------
__global__ void __launch_bounds__(128) agg_wmma(
    const float* __restrict__ h, const float* __restrict__ ds,
    const float* __restrict__ wf, float* __restrict__ agg,
    int Ks, int Cout)
{
  __shared__ _Float16 Hs[256 * GSTR];   // [c][k-chunk] transposed
  __shared__ _Float16 Ws[16 * GSTR];    // [w][k-chunk] transposed
  const int bs   = blockIdx.x;
  const int tid  = threadIdx.x;
  const int wave = tid >> 5, lane = tid & 31;
  const int lrow = lane & 15, lhi = lane >> 4;
  const size_t hbase = (size_t)bs * Ks * Cout;
  const size_t dbase = (size_t)bs * Ks;
  const size_t wbase = (size_t)bs * Ks * 16;
  const int ntiles = Cout >> 4;

  v8f acc[4] = {};

  for (int kb = 0; kb < Ks; kb += 32) {
    for (int idx = tid; idx < Cout * 32; idx += 128) {
      const int kk = idx / Cout, c = idx - kk * Cout;
      const int k = kb + kk;
      Hs[c * GSTR + kk] = (_Float16)(h[hbase + (size_t)k * Cout + c] * ds[dbase + k]);
    }
    for (int idx = tid; idx < 16 * 32; idx += 128) {
      const int kk = idx >> 4, w = idx & 15;
      Ws[w * GSTR + kk] = (_Float16)wf[wbase + (size_t)(kb + kk) * 16 + w];
    }
    __syncthreads();

    const int k0l = lhi ? 8 : 0;
    const int bb  = lrow * GSTR + (lhi ? 8 : 0);
    v16h bf = frag16(&Ws[bb], &Ws[bb + 16]);

#pragma unroll
    for (int t = 0; t < 4; ++t) {
      const int mt = wave + t * 4;
      if (mt < ntiles) {            // wave-uniform branch (EXEC stays all-1)
        const int ab = (mt * 16 + lrow) * GSTR;
        v16h af = frag16(&Hs[ab + k0l], &Hs[ab + 16 + k0l]);
        acc[t] = __builtin_amdgcn_wmma_f32_16x16x32_f16(false, af, false, bf,
                                                        (short)0, acc[t], false, false);
      }
    }
    __syncthreads();
  }

  float* ao = agg + (size_t)bs * Cout * 16;
#pragma unroll
  for (int t = 0; t < 4; ++t) {
    const int mt = wave + t * 4;
    if (mt >= ntiles) break;
#pragma unroll
    for (int i = 0; i < 8; ++i) {
      const int c = mt * 16 + lhi * 8 + i;
      ao[c * 16 + lrow] = acc[t][i];
    }
  }
}

// ---------------------------------------------------------------------------
// inv density: invd[b,n] = 2.5*bw*N / sum_m exp(-|p_n-p_m|^2/(2 bw^2))
// ---------------------------------------------------------------------------
__global__ void density_kernel(const float* __restrict__ xyz, int layout,
                               float* __restrict__ invd, int N, float bw)
{
  const int gid = blockIdx.x * blockDim.x + threadIdx.x;
  const int b = gid / N, n = gid % N;
  float x, y, z; load_pt(xyz, layout, b, n, N, x, y, z);
  const float inv2 = 1.f / (2.f * bw * bw);
  float s = 0.f;
  for (int m = 0; m < N; ++m) {
    float mx, my, mz; load_pt(xyz, layout, b, m, N, mx, my, mz);
    const float dx = x - mx, dy = y - my, dz = z - mz;
    s += expf(-(dx * dx + dy * dy + dz * dz) * inv2);
  }
  invd[gid] = (2.5f * bw * (float)N) / s;
}

// ---------------------------------------------------------------------------
// Farthest point sampling: one workgroup per batch, points+dist in LDS
// ---------------------------------------------------------------------------
__global__ void __launch_bounds__(256) fps_kernel(const float* __restrict__ xyz, int layout,
                                                  int* __restrict__ out, int N, int S)
{
  __shared__ float px[2048], py[2048], pz[2048], dist[2048];
  __shared__ float rv[256];
  __shared__ int   ri[256];
  __shared__ int   s_far;
  const int b = blockIdx.x, tid = threadIdx.x;
  for (int n = tid; n < N; n += 256) {
    float x, y, z; load_pt(xyz, layout, b, n, N, x, y, z);
    px[n] = x; py[n] = y; pz[n] = z; dist[n] = 1e10f;
  }
  if (tid == 0) s_far = 0;
  __syncthreads();
  for (int j = 0; j < S; ++j) {
    const int far = s_far;
    if (tid == 0) out[b * S + j] = far;
    const float cx = px[far], cy = py[far], cz = pz[far];
    float lbest = -1.f; int lidx = 0;
    for (int n = tid; n < N; n += 256) {
      const float dx = px[n] - cx, dy = py[n] - cy, dz = pz[n] - cz;
      const float dm = fminf(dist[n], dx * dx + dy * dy + dz * dz);
      dist[n] = dm;
      if (dm > lbest) { lbest = dm; lidx = n; }
    }
    rv[tid] = lbest; ri[tid] = lidx;
    __syncthreads();
    for (int off = 128; off > 0; off >>= 1) {
      if (tid < off) {
        if (rv[tid + off] > rv[tid] ||
            (rv[tid + off] == rv[tid] && ri[tid + off] < ri[tid])) {
          rv[tid] = rv[tid + off]; ri[tid] = ri[tid + off];
        }
      }
      __syncthreads();
    }
    if (tid == 0) s_far = ri[0];
    __syncthreads();
  }
}

__global__ void gather_newxyz(const float* __restrict__ xyz, int layout,
                              const int* __restrict__ fidx, float* __restrict__ nxyz,
                              int S, int N, int total)
{
  const int gid = blockIdx.x * blockDim.x + threadIdx.x;
  if (gid >= total) return;
  const int b = gid / S;
  float x, y, z; load_pt(xyz, layout, b, fidx[gid], N, x, y, z);
  nxyz[gid * 3 + 0] = x; nxyz[gid * 3 + 1] = y; nxyz[gid * 3 + 2] = z;
}

// ---------------------------------------------------------------------------
// kNN: one workgroup per (b,s); iterative K-way argmin (matches top_k order)
// ---------------------------------------------------------------------------
__global__ void __launch_bounds__(128) knn_kernel(const float* __restrict__ xyz, int layout,
                                                  const float* __restrict__ nxyz,
                                                  int* __restrict__ idxo,
                                                  int N, int S, int K)
{
  __shared__ float dist[2048];
  __shared__ float rv[128];
  __shared__ int   ri[128];
  const int bs = blockIdx.x, b = bs / S, tid = threadIdx.x;
  const float cx = nxyz[bs * 3], cy = nxyz[bs * 3 + 1], cz = nxyz[bs * 3 + 2];
  for (int n = tid; n < N; n += 128) {
    float x, y, z; load_pt(xyz, layout, b, n, N, x, y, z);
    const float dx = x - cx, dy = y - cy, dz = z - cz;
    dist[n] = dx * dx + dy * dy + dz * dz;
  }
  __syncthreads();
  for (int k = 0; k < K; ++k) {
    float lbest = 1e30f; int lidx = 0;
    for (int n = tid; n < N; n += 128)
      if (dist[n] < lbest) { lbest = dist[n]; lidx = n; }
    rv[tid] = lbest; ri[tid] = lidx;
    __syncthreads();
    for (int off = 64; off > 0; off >>= 1) {
      if (tid < off) {
        if (rv[tid + off] < rv[tid] ||
            (rv[tid + off] == rv[tid] && ri[tid + off] < ri[tid])) {
          rv[tid] = rv[tid + off]; ri[tid] = ri[tid + off];
        }
      }
      __syncthreads();
    }
    if (tid == 0) { idxo[(size_t)bs * K + k] = ri[0]; dist[ri[0]] = 1e30f; }
    __syncthreads();
  }
}

// ---------------------------------------------------------------------------
// build new_points (centered xyz ++ gathered points) and grouped density
// ---------------------------------------------------------------------------
__global__ void group_kernel(const float* __restrict__ xyz, int xlay,
                             const float* __restrict__ pts, int play, int Cpts,
                             const float* __restrict__ nxyz, const int* __restrict__ idx,
                             const float* __restrict__ invd,
                             float* __restrict__ np_, float* __restrict__ gd,
                             int S, int K, int N, int total)
{
  const int gid = blockIdx.x * blockDim.x + threadIdx.x;
  if (gid >= total) return;
  const int bs = gid / K, b = bs / S;
  const int i = idx[gid];
  const int cin = 3 + Cpts;
  float* o = np_ + (size_t)gid * cin;
  float x, y, z; load_pt(xyz, xlay, b, i, N, x, y, z);
  o[0] = x - nxyz[bs * 3];
  o[1] = y - nxyz[bs * 3 + 1];
  o[2] = z - nxyz[bs * 3 + 2];
  if (play == 1) {
    float a0, a1, a2; load_pt(pts, 1, b, i, N, a0, a1, a2);
    o[3] = a0; o[4] = a1; o[5] = a2;
  } else {
    const float* p = pts + ((size_t)b * N + i) * Cpts;
    for (int c = 0; c < Cpts; ++c) o[3 + c] = p[c];
  }
  gd[gid] = invd[(size_t)b * N + i];
}

__global__ void group_all_kernel(const float* __restrict__ xyz,
                                 const float* __restrict__ pts, int Cpts,
                                 const float* __restrict__ invd,
                                 float* __restrict__ np_, float* __restrict__ gd,
                                 int total)
{
  const int gid = blockIdx.x * blockDim.x + threadIdx.x;
  if (gid >= total) return;
  const int cin = 3 + Cpts;
  float* o = np_ + (size_t)gid * cin;
  const float* p3 = xyz + (size_t)gid * 3;
  o[0] = p3[0]; o[1] = p3[1]; o[2] = p3[2];
  const float* p = pts + (size_t)gid * Cpts;
  for (int c = 0; c < Cpts; ++c) o[3 + c] = p[c];
  gd[gid] = invd[gid];
}

// ---------------------------------------------------------------------------
// density-net: ds = dn3(dn2(dn1(gd / max_k gd)))   (1->16->8->1, affine+relu)
// one workgroup (K threads) per (b,s)
// ---------------------------------------------------------------------------
__global__ void ds_kernel(const float* __restrict__ gd, float* __restrict__ ds,
                          const float* W1, const float* b1, const float* g1, const float* e1,
                          const float* W2, const float* b2, const float* g2, const float* e2,
                          const float* W3, const float* b3, const float* g3, const float* e3,
                          int K)
{
  __shared__ float red[128];
  const int bs = blockIdx.x, tid = threadIdx.x;
  const float v = gd[(size_t)bs * K + tid];
  red[tid] = v;
  __syncthreads();
  for (int off = K >> 1; off > 0; off >>= 1) {
    if (tid < off) red[tid] = fmaxf(red[tid], red[tid + off]);
    __syncthreads();
  }
  const float x0 = v / red[0];
  float h1[16];
#pragma unroll
  for (int o = 0; o < 16; ++o)
    h1[o] = fmaxf((W1[o] * x0 + b1[o]) * g1[o] + e1[o], 0.f);
  float h2[8];
#pragma unroll
  for (int o = 0; o < 8; ++o) {
    float a = b2[o];
#pragma unroll
    for (int c = 0; c < 16; ++c) a += W2[o * 16 + c] * h1[c];
    h2[o] = fmaxf((a) * g2[o] + e2[o], 0.f);
  }
  float a = b3[0];
#pragma unroll
  for (int c = 0; c < 8; ++c) a += W3[c] * h2[c];
  ds[(size_t)bs * K + tid] = fmaxf(a * g3[0] + e3[0], 0.f);
}

// ---------------------------------------------------------------------------
// weight-net: wf = wn3(wn2(wn1(grouped_xyz)))   (3->8->8->16, affine+relu)
// grouped_xyz = first 3 channels of new_points
// ---------------------------------------------------------------------------
__global__ void wn_kernel(const float* __restrict__ np_, int cin, float* __restrict__ wf,
                          const float* W1, const float* b1, const float* g1, const float* e1,
                          const float* W2, const float* b2, const float* g2, const float* e2,
                          const float* W3, const float* b3, const float* g3, const float* e3,
                          int total)
{
  const int gid = blockIdx.x * blockDim.x + threadIdx.x;
  if (gid >= total) return;
  const float* p = np_ + (size_t)gid * cin;
  const float x = p[0], y = p[1], z = p[2];
  float h1[8];
#pragma unroll
  for (int o = 0; o < 8; ++o)
    h1[o] = fmaxf((W1[o * 3] * x + W1[o * 3 + 1] * y + W1[o * 3 + 2] * z + b1[o]) * g1[o] + e1[o], 0.f);
  float h2[8];
#pragma unroll
  for (int o = 0; o < 8; ++o) {
    float a = b2[o];
#pragma unroll
    for (int c = 0; c < 8; ++c) a += W2[o * 8 + c] * h1[c];
    h2[o] = fmaxf(a * g2[o] + e2[o], 0.f);
  }
  float* out = wf + (size_t)gid * 16;
#pragma unroll
  for (int o = 0; o < 16; ++o) {
    float a = b3[o];
#pragma unroll
    for (int c = 0; c < 8; ++c) a += W3[o * 8 + c] * h2[c];
    out[o] = fmaxf(a * g3[o] + e3[o], 0.f);
  }
}

// ---------------------------------------------------------------------------
// groupnorm over last dim (in-place) + optional relu
// ---------------------------------------------------------------------------
__global__ void __launch_bounds__(256) gn_kernel(float* __restrict__ x,
                                                 const float* __restrict__ g,
                                                 const float* __restrict__ b_,
                                                 int L, int relu)
{
  __shared__ float rs[256], rq[256];
  const int b = blockIdx.x, tid = threadIdx.x;
  float* row = x + (size_t)b * L;
  float s = 0.f, q = 0.f;
  for (int i = tid; i < L; i += 256) { const float v = row[i]; s += v; q += v * v; }
  rs[tid] = s; rq[tid] = q;
  __syncthreads();
  for (int off = 128; off > 0; off >>= 1) {
    if (tid < off) { rs[tid] += rs[tid + off]; rq[tid] += rq[tid + off]; }
    __syncthreads();
  }
  const float mu  = rs[0] / (float)L;
  const float var = rq[0] / (float)L - mu * mu;
  const float inv = rsqrtf(var + 1e-5f);
  for (int i = tid; i < L; i += 256) {
    float v = (row[i] - mu) * inv * g[i] + b_[i];
    if (relu) v = fmaxf(v, 0.f);
    row[i] = v;
  }
}

// log_softmax over axis=1 of (B,2,1024)
__global__ void lsm_kernel(const float* __restrict__ h, float* __restrict__ out)
{
  const int gid = blockIdx.x * blockDim.x + threadIdx.x;   // 8*1024
  const int b = gid >> 10, j = gid & 1023;
  const float a = h[b * 2048 + j], c = h[b * 2048 + 1024 + j];
  const float m = fmaxf(a, c);
  const float l = m + logf(expf(a - m) + expf(c - m));
  out[b * 2048 + j]        = a - l;
  out[b * 2048 + 1024 + j] = c - l;
}

__global__ void copy_feat(const float* __restrict__ src, float* __restrict__ feat, int br)
{
  const int gid = blockIdx.x * blockDim.x + threadIdx.x;   // 8*256
  const int b = gid >> 8, c = gid & 255;
  feat[b * 512 + br * 256 + c] = src[gid];
}

// ---------------------------------------------------------------------------
// host side
// ---------------------------------------------------------------------------
extern "C" void kernel_launch(void* const* d_in, const int* in_sizes, int n_in,
                              void* d_out, int out_size, void* d_ws, size_t ws_size,
                              hipStream_t stream)
{
  (void)in_sizes; (void)n_in; (void)out_size; (void)ws_size;
  const int B = 8;
  auto fp = [&](int i) { return (const float*)d_in[i]; };

  // workspace carve-up (floats)
  float* ws = (float*)d_ws;
  size_t off = 0;
  auto alloc = [&](size_t n) { float* p = ws + off; off += n; return p; };
  float* invd   = alloc(16384);          // (8,2048)
  int*   fpsidx = (int*)alloc(4096);     // (8,512)
  float* nxyzA  = alloc(12288);          // (8,512,3)
  float* nxyzB  = alloc(3072);           // (8,128,3)
  int*   idx    = (int*)alloc(131072);   // (8,512,32) max
  float* npts   = alloc(4390912);        // (8,128,64,67) max
  float* gd     = alloc(131072);
  float* dsb    = alloc(131072);
  float* hb     = alloc(8388608);        // (8,512,32,64) max
  float* wfb    = alloc(2097152);        // (8,512,32,16) max
  float* aggb   = alloc(4194304);        // (8,512,1024) max
  float* outA   = alloc(262144);         // (8,512,64) max
  float* outB   = alloc(262144);
  float* feat   = alloc(4096);           // (8,512)
  float* h1     = alloc(4096);           // (8,512)
  float* h4     = alloc(8192);           // (8,1024)
  float* h5     = alloc(16384);          // (8,2048)

  // SA tail: mlp-GEMM -> dn -> wn -> agg-WMMA -> lin-GEMM
  auto run_tail = [&](int pb, int S, int Kk, int cin, int cout, float* outbuf) {
    const int M = B * S * Kk;
    dim3 gm((unsigned)((cout + 63) / 64), (unsigned)((M + 63) / 64));
    wmma_gemm<<<gm, 256, 0, stream>>>(npts, fp(pb + 0), fp(pb + 1), fp(pb + 2), fp(pb + 3),
                                      hb, M, cout, cin, /*bias|affine|relu*/7);
    ds_kernel<<<B * S, Kk, 0, stream>>>(gd, dsb,
        fp(pb + 16), fp(pb + 17), fp(pb + 18), fp(pb + 19),
        fp(pb + 20), fp(pb + 21), fp(pb + 22), fp(pb + 23),
        fp(pb + 24), fp(pb + 25), fp(pb + 26), fp(pb + 27), Kk);
    wn_kernel<<<(M + 255) / 256, 256, 0, stream>>>(npts, cin, wfb,
        fp(pb + 4),  fp(pb + 5),  fp(pb + 6),  fp(pb + 7),
        fp(pb + 8),  fp(pb + 9),  fp(pb + 10), fp(pb + 11),
        fp(pb + 12), fp(pb + 13), fp(pb + 14), fp(pb + 15), M);
    agg_wmma<<<B * S, 128, 0, stream>>>(hb, dsb, wfb, aggb, Kk, cout);
    dim3 gl((unsigned)((cout + 63) / 64), (unsigned)((B * S + 63) / 64));
    wmma_gemm<<<gl, 256, 0, stream>>>(aggb, fp(pb + 28), fp(pb + 29), fp(pb + 30), fp(pb + 31),
                                      outbuf, B * S, cout, 16 * cout, 7);
  };

  auto run_branch = [&](const float* cloud, int br) {
    // ---------------- SA1: N=2048, S=512, K=32, cin=6, cout=64 --------------
    {
      const int N = 2048, S = 512, Kk = 32;
      density_kernel<<<B * N / 256, 256, 0, stream>>>(cloud, 1, invd, N, 0.1f);
      fps_kernel<<<B, 256, 0, stream>>>(cloud, 1, fpsidx, N, S);
      gather_newxyz<<<(B * S + 255) / 256, 256, 0, stream>>>(cloud, 1, fpsidx, nxyzA, S, N, B * S);
      knn_kernel<<<B * S, 128, 0, stream>>>(cloud, 1, nxyzA, idx, N, S, Kk);
      group_kernel<<<(B * S * Kk + 255) / 256, 256, 0, stream>>>(
          cloud, 1, cloud, 1, 3, nxyzA, idx, invd, npts, gd, S, Kk, N, B * S * Kk);
      run_tail(2, S, Kk, 6, 64, outA);
    }
    // ---------------- SA2: N=512, S=128, K=64, cin=67, cout=128 -------------
    {
      const int N = 512, S = 128, Kk = 64;
      density_kernel<<<B * N / 256, 256, 0, stream>>>(nxyzA, 0, invd, N, 0.2f);
      fps_kernel<<<B, 256, 0, stream>>>(nxyzA, 0, fpsidx, N, S);
      gather_newxyz<<<(B * S + 255) / 256, 256, 0, stream>>>(nxyzA, 0, fpsidx, nxyzB, S, N, B * S);
      knn_kernel<<<B * S, 128, 0, stream>>>(nxyzA, 0, nxyzB, idx, N, S, Kk);
      group_kernel<<<(B * S * Kk + 255) / 256, 256, 0, stream>>>(
          nxyzA, 0, outA, 0, 64, nxyzB, idx, invd, npts, gd, S, Kk, N, B * S * Kk);
      run_tail(34, S, Kk, 67, 128, outB);
    }
    // ---------------- SA3 (group_all): N=128, S=1, K=128, cin=131, cout=256 -
    {
      const int N = 128, Kk = 128;
      density_kernel<<<B * N / 256, 256, 0, stream>>>(nxyzB, 0, invd, N, 0.4f);
      group_all_kernel<<<(B * N + 255) / 256, 256, 0, stream>>>(
          nxyzB, outB, 128, invd, npts, gd, B * N);
      run_tail(66, 1, Kk, 131, 256, outA);
    }
    copy_feat<<<(B * 256 + 255) / 256, 256, 0, stream>>>(outA, feat, br);
  };

  run_branch(fp(0), 0);   // xyz branch
  run_branch(fp(1), 1);   // xyz_goal branch

  // ---------------- head: fc1+gn+relu, fc4+gn+relu, fc5, log_softmax --------
  {
    dim3 g1((512 + 63) / 64, 1);
    wmma_gemm<<<g1, 256, 0, stream>>>(feat, fp(98), fp(99), nullptr, nullptr, h1, B, 512, 512, 4);
    gn_kernel<<<B, 256, 0, stream>>>(h1, fp(100), fp(101), 512, 1);
    dim3 g4((1024 + 63) / 64, 1);
    wmma_gemm<<<g4, 256, 0, stream>>>(h1, fp(102), fp(103), nullptr, nullptr, h4, B, 1024, 512, 4);
    gn_kernel<<<B, 256, 0, stream>>>(h4, fp(104), fp(105), 1024, 1);
    dim3 g5((2048 + 63) / 64, 1);
    wmma_gemm<<<g5, 256, 0, stream>>>(h4, fp(106), fp(107), nullptr, nullptr, h5, B, 2048, 1024, 4);
    lsm_kernel<<<(B * 1024) / 256, 256, 0, stream>>>(h5, (float*)d_out);
  }
}
